// DGP_RF_Embeddings_14018773254666
// MI455X (gfx1250) — matrix operand order repaired and checked by reading, not compiled, not run on hardware
//
#include <hip/hip_runtime.h>
#include <hip/hip_bf16.h>
#include <math.h>

typedef __bf16 bf16;
typedef __attribute__((ext_vector_type(16))) __bf16 v16bf;
typedef __attribute__((ext_vector_type(8)))  __bf16 v8bf;
typedef __attribute__((ext_vector_type(8)))  float  v8f;
typedef __attribute__((ext_vector_type(4)))  float  v4f;

#define NROWS   131072
#define D_IN    128
#define D_RF    1024
#define D_OUT   128
#define NGROUP  4096
#define EPSF    1e-8f
#define RPB     32          // rows per block
#define LPITCH  1032        // bf16 elems per LDS row (1024 + 8 pad, keeps 16B alignment)

__device__ __forceinline__ v8f wmma_bf16(v16bf a, v16bf b, v8f c) {
    return __builtin_amdgcn_wmma_f32_16x16x32_bf16(false, a, false, b, (short)0, c, false, false);
}

// ---------------------------------------------------------------- prep: cast/exponentiate weights
__global__ __launch_bounds__(256)
void prep_kernel(const float* __restrict__ w0mu_f, const float* __restrict__ w0lv,
                 const float* __restrict__ b0lv,
                 const float* __restrict__ w1mu_f, const float* __restrict__ w1lv,
                 const float* __restrict__ b1lv,
                 bf16* __restrict__ w0mu, bf16* __restrict__ vw0,
                 bf16* __restrict__ w1mu, bf16* __restrict__ vw1, bf16* __restrict__ w1sq,
                 float* __restrict__ eb0, float* __restrict__ eb1)
{
    int i = blockIdx.x * blockDim.x + threadIdx.x;   // 0 .. 131071 (both weight mats have 1024*128)
    w0mu[i] = (bf16)w0mu_f[i];
    vw0[i]  = (bf16)__expf(w0lv[i]);
    float w1 = w1mu_f[i];
    w1mu[i] = (bf16)w1;
    vw1[i]  = (bf16)__expf(w1lv[i]);
    w1sq[i] = (bf16)(w1 * w1);
    if (i < D_RF)  eb0[i] = __expf(b0lv[i]);
    if (i < D_OUT) eb1[i] = __expf(b1lv[i]);
}

// ---------------------------------------------------------------- zero segment accumulators
__global__ __launch_bounds__(512)
void zero_kernel(float* __restrict__ p, int n)
{
    int i = blockIdx.x * blockDim.x + threadIdx.x;
    if (i < n) p[i] = 0.0f;
}

// ---------------------------------------------------------------- fused VB network + segment atomics
__global__ __launch_bounds__(256)
void vb_fused(const float* __restrict__ X, const int* __restrict__ Xidx,
              const float* __restrict__ b0_mu, const float* __restrict__ b1_mu,
              const bf16* __restrict__ w0mu, const bf16* __restrict__ vw0,
              const bf16* __restrict__ w1mu, const bf16* __restrict__ vw1,
              const bf16* __restrict__ w1sq,
              const float* __restrict__ eb0, const float* __restrict__ eb1,
              float* __restrict__ accwm)
{
    __shared__ __attribute__((aligned(16))) bf16 lm[RPB * LPITCH];   // E[relu]
    __shared__ __attribute__((aligned(16))) bf16 lv[RPB * LPITCH];   // Var[relu]
    __shared__ __attribute__((aligned(16))) bf16 lx2[RPB * LPITCH];  // E[relu^2] = e^2 + v

    const int lane = threadIdx.x & 31;
    const int wave = threadIdx.x >> 5;     // 8 waves
    const int hi   = lane >> 4;            // half-wave select
    const int ln   = lane & 15;
    const int row_base = blockIdx.x * RPB;

    // ================= layer 0: [32,128] x [128,1024] -> m1,v1 (ReLU moment prop) =================
    for (int mt = 0; mt < 2; ++mt) {
        // Cache A operands for all 4 K-steps: a_m = bf16(x), a_x2 = bf16(x*x)
        // A layout (16-bit, 16x32): lane M = ln; K = kk*32 + hi*16 + 0..15 (contiguous).
        v16bf a_m[4], a_x2[4];
        const float* xr = X + (long)(row_base + mt * 16 + ln) * D_IN;
        #pragma unroll
        for (int kk = 0; kk < 4; ++kk) {
            const int kb = kk * 32 + hi * 16;
            v4f x0 = *(const v4f*)(xr + kb);
            v4f x1 = *(const v4f*)(xr + kb + 4);
            v4f x2 = *(const v4f*)(xr + kb + 8);
            v4f x3 = *(const v4f*)(xr + kb + 12);
            #pragma unroll
            for (int i = 0; i < 4; ++i) {
                a_m[kk][i]      = (bf16)x0[i];  a_x2[kk][i]      = (bf16)(x0[i] * x0[i]);
                a_m[kk][4 + i]  = (bf16)x1[i];  a_x2[kk][4 + i]  = (bf16)(x1[i] * x1[i]);
                a_m[kk][8 + i]  = (bf16)x2[i];  a_x2[kk][8 + i]  = (bf16)(x2[i] * x2[i]);
                a_m[kk][12 + i] = (bf16)x3[i];  a_x2[kk][12 + i] = (bf16)(x3[i] * x3[i]);
            }
        }
        // 8 output-column tiles per wave (covers 1024 cols across 8 waves)
        for (int nj = 0; nj < 8; ++nj) {
            const int ncol = (wave + nj * 8) * 16 + ln;   // B lane column
            v8f cm = {}; v8f cv = {};
            #pragma unroll
            for (int kk = 0; kk < 4; ++kk) {
                const int kb = kk * 32 + hi * 16;         // B lane K chunk
                v16bf bmu = *(const v16bf*)(w0mu + ncol * D_IN + kb);
                v16bf bvw = *(const v16bf*)(vw0  + ncol * D_IN + kb);
                cm = wmma_bf16(a_m[kk],  bmu, cm);
                cv = wmma_bf16(a_x2[kk], bvw, cv);
            }
            const float bm = b0_mu[ncol];
            const float bv = eb0[ncol];
            #pragma unroll
            for (int r = 0; r < 8; ++r) {
                float m  = cm[r] + bm;
                float vp = cv[r] + bv + EPSF;
                // ReLU moment propagation — fully branchless.
                float rs  = __builtin_amdgcn_rsqf(vp);            // 1/sqrt(v+eps)
                float s   = vp * rs;                               // sqrt(v+eps)
                float al  = m * rs;
                float pdf = __expf(fmaf(-0.5f * al, al, -0.9189385332046727f)); // phi(al)
                // Normal CDF via Zelen–Severo branchless expansion (|err| < 7.5e-8):
                // Phi(a) = 1 - phi(a)*P(t), t = 1/(1+0.2316419|a|), reflected for a<0.
                float aabs = fabsf(al);
                float t  = __builtin_amdgcn_rcpf(fmaf(0.2316419f, aabs, 1.0f));
                float p5 = fmaf(t, 1.330274429f, -1.821255978f);
                p5 = fmaf(t, p5, 1.781477937f);
                p5 = fmaf(t, p5, -0.356563782f);
                p5 = fmaf(t, p5, 0.319381530f);
                float q   = pdf * (p5 * t);                        // upper-tail prob of |al|
                float cdf = (al >= 0.0f) ? (1.0f - q) : q;
                float spdf = s * pdf;
                float e   = fmaf(m, cdf, spdf);
                float vr  = fmaxf(fmaf(fmaf(m, m, vp), cdf, m * spdf) - e * e, EPSF);
                float x2  = fmaf(e, e, vr);                        // E[x^2] for next layer
                const int mrow = mt * 16 + r + hi * 8;             // C layout: M = r + 8*hi
                lm [mrow * LPITCH + ncol] = (bf16)e;
                lv [mrow * LPITCH + ncol] = (bf16)vr;
                lx2[mrow * LPITCH + ncol] = (bf16)x2;
            }
        }
    }
    __syncthreads();

    // ================= layer 1: [32,1024] x [1024,128] -> m2,v2; wave owns 16-col tile =================
    const int col = wave * 16 + ln;                 // output column 0..127
    v8f cm0 = {}, cv0 = {}, cm1 = {}, cv1 = {};
    for (int kk = 0; kk < 32; ++kk) {
        const int kbB = kk * 32 + hi * 16;
        v16bf bmu = *(const v16bf*)(w1mu + col * D_RF + kbB);
        v16bf bvw = *(const v16bf*)(vw1  + col * D_RF + kbB);
        v16bf bsq = *(const v16bf*)(w1sq + col * D_RF + kbB);
        #pragma unroll
        for (int mt = 0; mt < 2; ++mt) {
            const int arow = mt * 16 + ln;
            const int ca   = kk * 32 + hi * 16;
            v8bf mA0 = *(const v8bf*)(lm  + arow * LPITCH + ca);
            v8bf mA1 = *(const v8bf*)(lm  + arow * LPITCH + ca + 8);
            v8bf vA0 = *(const v8bf*)(lv  + arow * LPITCH + ca);
            v8bf vA1 = *(const v8bf*)(lv  + arow * LPITCH + ca + 8);
            v8bf xA0 = *(const v8bf*)(lx2 + arow * LPITCH + ca);
            v8bf xA1 = *(const v8bf*)(lx2 + arow * LPITCH + ca + 8);
            v16bf am, av, ax;
            #pragma unroll
            for (int i = 0; i < 8; ++i) {
                am[i] = mA0[i];  am[8 + i] = mA1[i];
                av[i] = vA0[i];  av[8 + i] = vA1[i];
                ax[i] = xA0[i];  ax[8 + i] = xA1[i];
            }
            if (mt == 0) {
                cm0 = wmma_bf16(am, bmu, cm0);
                cv0 = wmma_bf16(ax, bvw, cv0);
                cv0 = wmma_bf16(av, bsq, cv0);
            } else {
                cm1 = wmma_bf16(am, bmu, cm1);
                cv1 = wmma_bf16(ax, bvw, cv1);
                cv1 = wmma_bf16(av, bsq, cv1);
            }
        }
    }

    // ================= precision-weighted segment atomics (interleaved [w,m] pairs) =================
    const float bm1 = b1_mu[col];
    const float bv1 = eb1[col];
    #pragma unroll
    for (int mt = 0; mt < 2; ++mt) {
        v8f cm = mt ? cm1 : cm0;
        v8f cv = mt ? cv1 : cv0;
        #pragma unroll
        for (int r = 0; r < 8; ++r) {
            const int grow = row_base + mt * 16 + r + hi * 8;
            const int g    = Xidx[grow];
            float m    = cm[r] + bm1;
            float prec = __builtin_amdgcn_rcpf(cv[r] + bv1 + EPSF);
            float* base = accwm + (((long)g * D_OUT + col) << 1);
            unsafeAtomicAdd(base,     prec);
            unsafeAtomicAdd(base + 1, prec * m);
        }
    }
}

// ---------------------------------------------------------------- finalize: mean & var outputs
__global__ __launch_bounds__(512)
void finalize_kernel(const float* __restrict__ accwm, float* __restrict__ out, int n)
{
    int i = blockIdx.x * blockDim.x + threadIdx.x;
    if (i < n) {
        float w  = accwm[2 * i];
        float mm = accwm[2 * i + 1];
        float var = __builtin_amdgcn_rcpf(w + EPSF);   // w_sum already includes +EPS here
        out[i]     = mm * var;
        out[n + i] = var;
    }
}

extern "C" void kernel_launch(void* const* d_in, const int* in_sizes, int n_in,
                              void* d_out, int out_size, void* d_ws, size_t ws_size,
                              hipStream_t stream)
{
    const float* X    = (const float*)d_in[0];
    const int*   Xidx = (const int*)  d_in[1];
    const float* W0mu = (const float*)d_in[2];
    const float* W0lv = (const float*)d_in[3];
    const float* b0mu = (const float*)d_in[4];
    const float* b0lv = (const float*)d_in[5];
    const float* W1mu = (const float*)d_in[6];
    const float* W1lv = (const float*)d_in[7];
    const float* b1mu = (const float*)d_in[8];
    const float* b1lv = (const float*)d_in[9];

    char* ws = (char*)d_ws;
    float* accwm  = (float*)(ws + 0);                  // 4096*128*2*4 = 4 MB interleaved [w,m]
    bf16*  w0mu_b = (bf16*) (ws + 4194304);            // 256 KB each
    bf16*  vw0_b  = (bf16*) (ws + 4456448);
    bf16*  w1mu_b = (bf16*) (ws + 4718592);
    bf16*  vw1_b  = (bf16*) (ws + 4980736);
    bf16*  w1sq_b = (bf16*) (ws + 5242880);
    float* eb0    = (float*)(ws + 5505024);
    float* eb1    = (float*)(ws + 5509120);

    zero_kernel<<<2048, 512, 0, stream>>>(accwm, 2 * NGROUP * D_OUT);
    prep_kernel<<<512, 256, 0, stream>>>(W0mu, W0lv, b0lv, W1mu, W1lv, b1lv,
                                         w0mu_b, vw0_b, w1mu_b, vw1_b, w1sq_b, eb0, eb1);
    vb_fused<<<NROWS / RPB, 256, 0, stream>>>(X, Xidx, b0mu, b1mu,
                                              w0mu_b, vw0_b, w1mu_b, vw1_b, w1sq_b,
                                              eb0, eb1, accwm);
    finalize_kernel<<<1024, 512, 0, stream>>>(accwm, (float*)d_out, NGROUP * D_OUT);
}
